// TransformerDecoderBlock_28020366639544
// MI455X (gfx1250) — matrix operand order repaired and checked by reading, not compiled, not run on hardware
//
#include <hip/hip_runtime.h>
#include <cmath>
#include <cstdint>

// ---------------- problem constants ----------------
namespace cfg {
constexpr int B = 4, S = 1024, D = 1024, H = 16, HD = 64;
constexpr int F = 4096, E = 8, TOPK = 2;
constexpr int T = B * S;                 // 4096 tokens
constexpr int CAP = T * TOPK + E * 128;  // padded slot capacity = 9216
constexpr int NT = CAP / 128;            // 72 row-tiles in slot space
}

typedef __attribute__((ext_vector_type(16))) __bf16 v16bf;
typedef __attribute__((ext_vector_type(8)))  __bf16 v8bf;
typedef __attribute__((ext_vector_type(8)))  float  v8f;
typedef __attribute__((ext_vector_type(4)))  unsigned int u32x4;
typedef __attribute__((ext_vector_type(8)))  int i32x8;
typedef __attribute__((ext_vector_type(4)))  int i32x4;

static __device__ __forceinline__ v16bf pack16(v8bf lo, v8bf hi) {
  v16bf f;
#pragma unroll
  for (int i = 0; i < 8; ++i) { f[i] = lo[i]; f[i + 8] = hi[i]; }
  return f;
}

static __device__ __forceinline__ void wait_dscnt0() {
  asm volatile("s_wait_dscnt 0x0" ::: "memory");
}

// ---- Tensor Data Mover: async 2D tile load Global -> LDS with row padding ----
// Loads a (tile_dim1=128 rows) x (tile_dim0=32 bf16) tile whose rows are lda
// elements apart in global memory, into LDS with 16 bytes of padding inserted
// after every 64 bytes (=> LDS row stride of 40 bf16, matching the WMMA
// fragment loader below). D# layout per CDNA5 ISA 8.3/8.4.
static __device__ __forceinline__ void tdm_load_A(const __bf16* gsrc,
                                                  unsigned lds_byte_off,
                                                  int Kdim, int Mdim, int lda) {
  const unsigned long long ga = (unsigned long long)(uintptr_t)gsrc;
  u32x4 g0;
  g0[0] = 1u;                                      // count=1 (valid user D#)
  g0[1] = lds_byte_off;                            // lds_addr
  g0[2] = (unsigned)(ga & 0xffffffffu);            // global_addr[31:0]
  g0[3] = (unsigned)((ga >> 32) & 0x01ffffffu)     // global_addr[56:32]
        | (2u << 30);                              // type=2 ("image")
  i32x8 g1;
  g1[0] = (1 << 16)                                // data_size = 2 bytes
        | (1 << 20)                                // pad_enable
        | (3 << 22)                                // pad_interval: 16 DWORDs
        | (3 << 25);                               // pad_amount: 4 DWORDs (16B)
  g1[1] = (Kdim & 0xffff) << 16;                   // tensor_dim0[15:0]
  g1[2] = ((unsigned)Kdim >> 16) | ((unsigned)(Mdim & 0xffff) << 16);
  g1[3] = (((unsigned)Mdim >> 16) & 0xffff) | (32u << 16);  // tile_dim0 = 32
  g1[4] = 128;                                     // tile_dim1 = 128, tile_dim2 = 0
  g1[5] = lda;                                     // tensor_dim0_stride[31:0]
  g1[6] = 0;
  g1[7] = 0;
  const i32x4 z4 = {0, 0, 0, 0};
#if defined(__clang_major__) && (__clang_major__ >= 23)
  const i32x8 z8 = {0, 0, 0, 0, 0, 0, 0, 0};
  __builtin_amdgcn_tensor_load_to_lds(g0, g1, z4, z4, z8, 0);
#else
  __builtin_amdgcn_tensor_load_to_lds(g0, g1, z4, z4, 0);
#endif
}

// A-fragment (16x32 bf16) from LDS row-major tile, row stride 40 elements.
static __device__ __forceinline__ v16bf lds_fragA(const __bf16* base, int lane) {
  const int sel = (lane >> 4) & 1, r = lane & 15;
  v8bf lo = *(const v8bf*)(base + r * 40 + sel * 8);
  v8bf hi = *(const v8bf*)(base + r * 40 + 16 + sel * 8);
  return pack16(lo, hi);
}

// B-fragment (32x16 bf16) from LDS tile stored K-contiguous per column.
static __device__ __forceinline__ v16bf lds_fragB(const __bf16* base, int lane) {
  const int sel = (lane >> 4) & 1, n = lane & 15;
  v8bf lo = *(const v8bf*)(base + n * 40 + sel * 8);
  v8bf hi = *(const v8bf*)(base + n * 40 + 16 + sel * 8);
  return pack16(lo, hi);
}

// ---------------- f32 -> bf16 convert ----------------
__global__ void k_cvt(const float* __restrict__ s, __bf16* __restrict__ d, int n) {
  for (int i = blockIdx.x * blockDim.x + threadIdx.x; i < n; i += gridDim.x * blockDim.x)
    d[i] = (__bf16)s[i];
}

// ---------------- zero fill (graph-capture safe) ----------------
__global__ void k_zero(unsigned* __restrict__ p, int n) {
  for (int i = blockIdx.x * blockDim.x + threadIdx.x; i < n; i += gridDim.x * blockDim.x)
    p[i] = 0u;
}

// ---------------- LayerNorm (per row of D=1024) -> bf16 ----------------
__global__ __launch_bounds__(256) void k_ln(const float* __restrict__ x,
                                            const float* __restrict__ g,
                                            const float* __restrict__ bta,
                                            __bf16* __restrict__ out) {
  const int row = blockIdx.x;
  const float* xr = x + (size_t)row * cfg::D;
  const int t4 = threadIdx.x * 4;
  float4 v = *(const float4*)(xr + t4);
  float s  = v.x + v.y + v.z + v.w;
  float s2 = v.x * v.x + v.y * v.y + v.z * v.z + v.w * v.w;
#pragma unroll
  for (int m = 16; m >= 1; m >>= 1) { s += __shfl_xor(s, m); s2 += __shfl_xor(s2, m); }
  __shared__ float a1[8], a2[8], mb[2];
  const int wid = threadIdx.x >> 5, lane = threadIdx.x & 31;
  if (lane == 0) { a1[wid] = s; a2[wid] = s2; }
  __syncthreads();
  if (threadIdx.x == 0) {
    float t1 = 0.f, t2 = 0.f;
#pragma unroll
    for (int i = 0; i < 8; ++i) { t1 += a1[i]; t2 += a2[i]; }
    const float mean = t1 * (1.0f / cfg::D);
    const float var  = t2 * (1.0f / cfg::D) - mean * mean;
    mb[0] = mean; mb[1] = rsqrtf(var + 1e-5f);
  }
  __syncthreads();
  const float mean = mb[0], rstd = mb[1];
  const float4 gg = *(const float4*)(g + t4);
  const float4 bb = *(const float4*)(bta + t4);
  __bf16* orow = out + (size_t)row * cfg::D;
  orow[t4 + 0] = (__bf16)((v.x - mean) * rstd * gg.x + bb.x);
  orow[t4 + 1] = (__bf16)((v.y - mean) * rstd * gg.y + bb.y);
  orow[t4 + 2] = (__bf16)((v.z - mean) * rstd * gg.z + bb.z);
  orow[t4 + 3] = (__bf16)((v.w - mean) * rstd * gg.w + bb.w);
}

// ---------------- generic WMMA bf16 GEMM ----------------
// Block: 256 threads = 8 waves (2 M-waves x 4 N-waves). Block tile 128x128,
// wave tile 64x32, K-step 32. A tile staged via TDM (USE_TDM) or manual
// (gathered rows for grouped MoE).
// MODE 0: C -> bf16 | MODE 1: C+resid -> f32 | MODE 2: gelu(C+bias) -> bf16
// MODE 3: (C+bias)*scale[row] atomically scattered to outF[outIdx[row]]
template <int MODE, bool USE_TDM>
__global__ __launch_bounds__(256) void k_gemm(
    const __bf16* __restrict__ A, int lda,
    const __bf16* __restrict__ Bw, int ldb, int K, int Mrows,
    const int* __restrict__ rowidx, const int* __restrict__ rowtile_e, long bStrideE,
    const float* __restrict__ bias, int biasStride,
    const float* __restrict__ resid, const float* __restrict__ scale,
    const int* __restrict__ outIdx,
    float* __restrict__ outF, __bf16* __restrict__ outB, int ldc) {
  __shared__ __attribute__((aligned(64))) __bf16 As[128 * 40];
  __shared__ __attribute__((aligned(64))) __bf16 Bt[128 * 40];

  const int m0 = blockIdx.x * 128;
  const int n0 = blockIdx.y * 128;
  if (rowtile_e) {
    const int e = rowtile_e[blockIdx.x];
    if (e < 0) return;
    Bw += (long)e * bStrideE;
    if (bias) bias += (long)e * biasStride;
  }
  const int tid = threadIdx.x;
  const int wid = tid >> 5, lane = tid & 31;
  const int wm = wid >> 2, wn = wid & 3;
  const int sel = lane >> 4, ln = lane & 15;
  const unsigned asOff = (unsigned)(uintptr_t)As;

  v8f acc[4][2] = {};

  for (int k0 = 0; k0 < K; k0 += 32) {
    __syncthreads();  // previous tile fully consumed
    if (USE_TDM) {
      if (wid == 0)   // one wave issues the async tensor DMA for the A tile
        tdm_load_A(A + (size_t)m0 * lda + k0, asOff, K, Mrows, lda);
    } else {
      // manual gathered staging: 128 rows x 32 cols, 16B chunks
#pragma unroll 2
      for (int idx = tid; idx < 128 * 4; idx += 256) {
        const int r = idx >> 2, c = (idx & 3) * 8;
        const int gr = rowidx[m0 + r];
        const __bf16* src = A + (size_t)gr * lda + k0 + c;
        *(uint4*)&As[r * 40 + c] = *(const uint4*)src;
        if (k0 + 32 < K) __builtin_prefetch(src + 32, 0, 1);
      }
    }
    // stage B transposed: coalesced 16B global loads, u16 LDS scatter
#pragma unroll 2
    for (int idx = tid; idx < 32 * 16; idx += 256) {
      const int k = idx >> 4, n8 = (idx & 15) * 8;
      const __bf16* src = Bw + (size_t)(k0 + k) * ldb + n0 + n8;
      const v8bf vv = *(const v8bf*)src;
      if (k0 + 32 < K) __builtin_prefetch(src + (size_t)32 * ldb, 0, 1);
#pragma unroll
      for (int j = 0; j < 8; ++j) Bt[(n8 + j) * 40 + k] = vv[j];
    }
    if (USE_TDM) {
      if (wid == 0) __builtin_amdgcn_s_wait_tensorcnt(0);
    }
    __syncthreads();  // publishes TDM-A and manual-B tiles

    v16bf bf[2];
#pragma unroll
    for (int ni = 0; ni < 2; ++ni)
      bf[ni] = lds_fragB(Bt + (wn * 32 + ni * 16) * 40, lane);
#pragma unroll
    for (int mi = 0; mi < 4; ++mi) {
      const v16bf af = lds_fragA(As + (wm * 64 + mi * 16) * 40, lane);
#pragma unroll
      for (int ni = 0; ni < 2; ++ni)
        acc[mi][ni] = __builtin_amdgcn_wmma_f32_16x16x32_bf16(
            false, af, false, bf[ni], (short)0, acc[mi][ni], false, false);
    }
  }

  // epilogue: C layout element (row = m0+wm*64+mi*16+i+8*sel, col = n0+wn*32+ni*16+ln)
#pragma unroll
  for (int mi = 0; mi < 4; ++mi)
#pragma unroll
    for (int ni = 0; ni < 2; ++ni)
#pragma unroll
      for (int i = 0; i < 8; ++i) {
        const int row = m0 + wm * 64 + mi * 16 + i + 8 * sel;
        const int col = n0 + wn * 32 + ni * 16 + ln;
        float v = acc[mi][ni][i];
        if (MODE == 0) {
          outB[(size_t)row * ldc + col] = (__bf16)v;
        } else if (MODE == 1) {
          outF[(size_t)row * ldc + col] = resid[(size_t)row * ldc + col] + v;
        } else if (MODE == 2) {
          v += bias[col];
          const float gl = 0.5f * v * (1.0f + erff(v * 0.70710678f));
          outB[(size_t)row * ldc + col] = (__bf16)gl;
        } else {
          v = (v + bias[col]) * scale[row];
          atomicAdd(outF + (size_t)outIdx[row] * ldc + col, v);
        }
      }
}

// ---------------- flash attention (causal), bf16 in/out, f32 accum ----------------
__global__ __launch_bounds__(256) void k_attn(const __bf16* __restrict__ q,
                                              const __bf16* __restrict__ kM,
                                              const __bf16* __restrict__ vM,
                                              __bf16* __restrict__ o) {
  __shared__ __attribute__((aligned(64))) __bf16 sP[8][16 * 32];
  __shared__ __attribute__((aligned(64))) __bf16 sV[8][64 * 32];

  const int wid = threadIdx.x >> 5, lane = threadIdx.x & 31;
  const int sel = lane >> 4, ln = lane & 15;
  const int bb = blockIdx.z, hh = blockIdx.y;
  const int q0 = blockIdx.x * 128 + wid * 16;
  const size_t rowBase = ((size_t)bb * cfg::S) * cfg::D + (size_t)hh * cfg::HD;

  v16bf aq[2];
  {
    const __bf16* qp = q + rowBase + (size_t)(q0 + ln) * cfg::D;
#pragma unroll
    for (int kk = 0; kk < 2; ++kk) {
      v8bf lo = *(const v8bf*)(qp + kk * 32 + sel * 8);
      v8bf hi = *(const v8bf*)(qp + kk * 32 + 16 + sel * 8);
      aq[kk] = pack16(lo, hi);
    }
  }

  float m[8], l[8];
  v8f oacc[4] = {};
#pragma unroll
  for (int i = 0; i < 8; ++i) { m[i] = -1e30f; l[i] = 0.f; }

  const int ktiles = (q0 + 16 + 31) / 32;
  for (int kt = 0; kt < ktiles; ++kt) {
    const int k0 = kt * 32;
    // hoist all four K-matrix B fragments (two 16B loads each) before WMMA
    v16bf bks[2][2];
#pragma unroll
    for (int c = 0; c < 2; ++c) {
      const __bf16* kp = kM + rowBase + (size_t)(k0 + c * 16 + ln) * cfg::D;
#pragma unroll
      for (int kk = 0; kk < 2; ++kk)
        bks[c][kk] = pack16(*(const v8bf*)(kp + kk * 32 + sel * 8), *(const v8bf*)(kp + kk * 32 + 16 + sel * 8));
    }
    v8f sc[2] = {};
#pragma unroll
    for (int c = 0; c < 2; ++c)
#pragma unroll
      for (int kk = 0; kk < 2; ++kk)
        sc[c] = __builtin_amdgcn_wmma_f32_16x16x32_bf16(false, aq[kk], false, bks[c][kk],
                                                        (short)0, sc[c], false, false);
    // online softmax update
#pragma unroll
    for (int i = 0; i < 8; ++i) {
      const int row = q0 + i + 8 * sel;
      float v0 = sc[0][i] * 0.125f; if (k0 + ln > row)      v0 = -1e30f;
      float v1 = sc[1][i] * 0.125f; if (k0 + 16 + ln > row) v1 = -1e30f;
      float mm = fmaxf(v0, v1);
#pragma unroll
      for (int msk = 8; msk >= 1; msk >>= 1) mm = fmaxf(mm, __shfl_xor(mm, msk));
      const float mn = fmaxf(m[i], mm);
      const float alpha = __expf(m[i] - mn);
      m[i] = mn;
      const float p0 = __expf(v0 - mn), p1 = __expf(v1 - mn);
      float ps = p0 + p1;
#pragma unroll
      for (int msk = 8; msk >= 1; msk >>= 1) ps += __shfl_xor(ps, msk);
      l[i] = l[i] * alpha + ps;
#pragma unroll
      for (int j = 0; j < 4; ++j) oacc[j][i] *= alpha;
      sP[wid][(i + 8 * sel) * 32 + ln]      = (__bf16)p0;
      sP[wid][(i + 8 * sel) * 32 + 16 + ln] = (__bf16)p1;
    }
    // V tile transposed into LDS: sV[hd][key]
    for (int idx = lane; idx < 2048; idx += 32) {
      const int hd = idx & 63, key = idx >> 6;
      sV[wid][hd * 32 + key] = vM[rowBase + (size_t)(k0 + key) * cfg::D + hd];
    }
    wait_dscnt0();  // same-wave LDS RAW: s_wait_dscnt 0

    v16bf ap;
    {
      const __bf16* pb = sP[wid] + ln * 32;
      ap = pack16(*(const v8bf*)(pb + sel * 8), *(const v8bf*)(pb + 16 + sel * 8));
    }
#pragma unroll
    for (int j = 0; j < 4; ++j) {
      const v16bf bv = pack16(*(const v8bf*)(sV[wid] + (j * 16 + ln) * 32 + sel * 8), *(const v8bf*)(sV[wid] + (j * 16 + ln) * 32 + 16 + sel * 8));
      oacc[j] = __builtin_amdgcn_wmma_f32_16x16x32_bf16(false, ap, false, bv,
                                                        (short)0, oacc[j], false, false);
    }
  }

#pragma unroll
  for (int i = 0; i < 8; ++i) {
    const int row = q0 + i + 8 * sel;
    const float inv = 1.0f / l[i];
#pragma unroll
    for (int j = 0; j < 4; ++j)
      o[rowBase + (size_t)row * cfg::D + j * 16 + ln] = (__bf16)(oacc[j][i] * inv);
  }
}

// ---------------- MoE gating: logits, top-2, softmax, counts ----------------
__global__ __launch_bounds__(256) void k_gate(const __bf16* __restrict__ h,
                                              const float* __restrict__ gw,
                                              const float* __restrict__ gb,
                                              int* __restrict__ tope,
                                              float* __restrict__ gsc,
                                              int* __restrict__ counts) {
  const int wid = threadIdx.x >> 5, lane = threadIdx.x & 31;
  const int t = blockIdx.x * 8 + wid;
  float a[cfg::E] = {};
  for (int d = lane; d < cfg::D; d += 32) {
    const float xv = (float)h[(size_t)t * cfg::D + d];
    const float* gr = gw + d * cfg::E;
#pragma unroll
    for (int e = 0; e < cfg::E; ++e) a[e] += xv * gr[e];
  }
#pragma unroll
  for (int e = 0; e < cfg::E; ++e)
#pragma unroll
    for (int msk = 16; msk >= 1; msk >>= 1) a[e] += __shfl_xor(a[e], msk);
  if (lane == 0) {
    int e0 = 0, e1 = -1;
#pragma unroll
    for (int e = 0; e < cfg::E; ++e) a[e] += gb[e];
#pragma unroll
    for (int e = 1; e < cfg::E; ++e) if (a[e] > a[e0]) e0 = e;
#pragma unroll
    for (int e = 0; e < cfg::E; ++e)
      if (e != e0 && (e1 < 0 || a[e] > a[e1])) e1 = e;
    const float p1 = __expf(a[e1] - a[e0]);
    const float inv = 1.0f / (1.0f + p1);
    tope[t * 2] = e0; tope[t * 2 + 1] = e1;
    gsc[t * 2] = inv; gsc[t * 2 + 1] = p1 * inv;
    atomicAdd(&counts[e0], 1);
    atomicAdd(&counts[e1], 1);
  }
}

// ---------------- slot-space scan: pad each expert to 128-row tiles ----------------
__global__ void k_scan(const int* __restrict__ counts, int* __restrict__ po,
                       int* __restrict__ cursors, int* __restrict__ rowtile_e) {
  if (threadIdx.x == 0 && blockIdx.x == 0) {
    int off = 0;
    for (int e = 0; e < cfg::E; ++e) {
      po[e] = off; cursors[e] = 0;
      off += (counts[e] + 127) & ~127;
    }
    po[cfg::E] = off;
    for (int tile = 0; tile < cfg::NT; ++tile) {
      const int base = tile * 128;
      int e = -1;
      if (base < off) { e = 0; while (e < cfg::E - 1 && base >= po[e + 1]) ++e; }
      rowtile_e[tile] = e;
    }
  }
}

// ---------------- scatter tokens into per-expert slot lists ----------------
__global__ void k_fill(const int* __restrict__ tope, const float* __restrict__ gsc,
                       const int* __restrict__ po, int* __restrict__ cursors,
                       int* __restrict__ slot_tok, float* __restrict__ slot_scale) {
  const int t = blockIdx.x * blockDim.x + threadIdx.x;
  if (t >= cfg::T) return;
#pragma unroll
  for (int kk = 0; kk < 2; ++kk) {
    const int e = tope[t * 2 + kk];
    const int pos = atomicAdd(&cursors[e], 1);
    const int slot = po[e] + pos;
    slot_tok[slot] = t;
    slot_scale[slot] = gsc[t * 2 + kk];
  }
}

// ---------------- final residual: out = x2 + moe ----------------
__global__ __launch_bounds__(256) void k_addres(const float* __restrict__ a,
                                                const float* __restrict__ bm,
                                                float* __restrict__ out, int n4) {
  const int i = blockIdx.x * blockDim.x + threadIdx.x;
  if (i >= n4) return;
  const float4 va = ((const float4*)a)[i];
  const float4 vb = ((const float4*)bm)[i];
  ((float4*)out)[i] = make_float4(va.x + vb.x, va.y + vb.y, va.z + vb.z, va.w + vb.w);
}

// ---------------- host orchestration ----------------
extern "C" void kernel_launch(void* const* d_in, const int* in_sizes, int n_in,
                              void* d_out, int out_size, void* d_ws, size_t ws_size,
                              hipStream_t stream) {
  using namespace cfg;
  const float* x     = (const float*)d_in[0];
  const float* ln1_g = (const float*)d_in[1];
  const float* ln1_b = (const float*)d_in[2];
  const float* wq    = (const float*)d_in[3];
  const float* wk    = (const float*)d_in[4];
  const float* wv    = (const float*)d_in[5];
  const float* wo    = (const float*)d_in[6];
  const float* ln2_g = (const float*)d_in[7];
  const float* ln2_b = (const float*)d_in[8];
  const float* gw    = (const float*)d_in[9];
  const float* gb    = (const float*)d_in[10];
  const float* w1    = (const float*)d_in[11];
  const float* b1    = (const float*)d_in[12];
  const float* w2    = (const float*)d_in[13];
  const float* b2    = (const float*)d_in[14];
  float* out = (float*)d_out;

  char* ws = (char*)d_ws;
  size_t off = 0;
  auto alloc = [&](size_t bytes) -> char* {
    char* p = ws + off;
    off = (off + bytes + 255) & ~(size_t)255;
    return p;
  };
  __bf16* wq_bf = (__bf16*)alloc((size_t)D * D * 2);
  __bf16* wk_bf = (__bf16*)alloc((size_t)D * D * 2);
  __bf16* wv_bf = (__bf16*)alloc((size_t)D * D * 2);
  __bf16* wo_bf = (__bf16*)alloc((size_t)D * D * 2);
  __bf16* w1_bf = (__bf16*)alloc((size_t)E * D * F * 2);
  __bf16* w2_bf = (__bf16*)alloc((size_t)E * F * D * 2);
  __bf16* h1_bf = (__bf16*)alloc((size_t)T * D * 2);
  __bf16* qb    = (__bf16*)alloc((size_t)T * D * 2);
  __bf16* kb    = (__bf16*)alloc((size_t)T * D * 2);
  __bf16* vb    = (__bf16*)alloc((size_t)T * D * 2);
  __bf16* attb  = (__bf16*)alloc((size_t)T * D * 2);
  float*  x2    = (float*) alloc((size_t)T * D * 4);
  __bf16* h2_bf = (__bf16*)alloc((size_t)T * D * 2);
  float*  moe   = (float*) alloc((size_t)T * D * 4);
  int*    tope  = (int*)   alloc((size_t)T * 2 * 4);
  float*  gsc   = (float*) alloc((size_t)T * 2 * 4);
  int*    counts   = (int*)  alloc(E * 4);
  int*    cursors  = (int*)  alloc(E * 4);
  int*    po       = (int*)  alloc((E + 1) * 4);
  int*    rtile_e  = (int*)  alloc(NT * 4);
  int*    slot_tok = (int*)  alloc(CAP * 4);
  float*  slot_sc  = (float*)alloc(CAP * 4);
  __bf16* hid_bf   = (__bf16*)alloc((size_t)CAP * F * 2);

  // 1) weights f32 -> bf16
  k_cvt<<<1024, 256, 0, stream>>>(wq, wq_bf, D * D);
  k_cvt<<<1024, 256, 0, stream>>>(wk, wk_bf, D * D);
  k_cvt<<<1024, 256, 0, stream>>>(wv, wv_bf, D * D);
  k_cvt<<<1024, 256, 0, stream>>>(wo, wo_bf, D * D);
  k_cvt<<<4096, 256, 0, stream>>>(w1, w1_bf, E * D * F);
  k_cvt<<<4096, 256, 0, stream>>>(w2, w2_bf, E * F * D);

  // 2) zero accumulators / routing state
  k_zero<<<2048, 256, 0, stream>>>((unsigned*)moe, T * D);
  k_zero<<<1, 32, 0, stream>>>((unsigned*)counts, E);
  k_zero<<<64, 256, 0, stream>>>((unsigned*)slot_tok, CAP);
  k_zero<<<64, 256, 0, stream>>>((unsigned*)slot_sc, CAP);

  // 3) LN1
  k_ln<<<T, 256, 0, stream>>>(x, ln1_g, ln1_b, h1_bf);

  // 4) QKV projections (WMMA + TDM A-staging)
  dim3 gQ(T / 128, D / 128);
  k_gemm<0, true><<<gQ, 256, 0, stream>>>(h1_bf, D, wq_bf, D, D, T, nullptr, nullptr, 0,
                                          nullptr, 0, nullptr, nullptr, nullptr,
                                          nullptr, qb, D);
  k_gemm<0, true><<<gQ, 256, 0, stream>>>(h1_bf, D, wk_bf, D, D, T, nullptr, nullptr, 0,
                                          nullptr, 0, nullptr, nullptr, nullptr,
                                          nullptr, kb, D);
  k_gemm<0, true><<<gQ, 256, 0, stream>>>(h1_bf, D, wv_bf, D, D, T, nullptr, nullptr, 0,
                                          nullptr, 0, nullptr, nullptr, nullptr,
                                          nullptr, vb, D);

  // 5) causal flash attention
  k_attn<<<dim3(S / 128, H, B), 256, 0, stream>>>(qb, kb, vb, attb);

  // 6) output projection + residual -> x2 (f32)
  k_gemm<1, true><<<gQ, 256, 0, stream>>>(attb, D, wo_bf, D, D, T, nullptr, nullptr, 0,
                                          nullptr, 0, x, nullptr, nullptr,
                                          x2, nullptr, D);

  // 7) LN2
  k_ln<<<T, 256, 0, stream>>>(x2, ln2_g, ln2_b, h2_bf);

  // 8) gate -> top-2 + routing
  k_gate<<<T / 8, 256, 0, stream>>>(h2_bf, gw, gb, tope, gsc, counts);
  k_scan<<<1, 32, 0, stream>>>(counts, po, cursors, rtile_e);
  k_fill<<<T / 256, 256, 0, stream>>>(tope, gsc, po, cursors, slot_tok, slot_sc);

  // 9) grouped FFN1 (gathered rows -> manual staging): hid = gelu(h2[tok] @ w1[e] + b1[e])
  k_gemm<2, false><<<dim3(NT, F / 128), 256, 0, stream>>>(
      h2_bf, D, w1_bf, F, D, CAP, slot_tok, rtile_e, (long)D * F,
      b1, F, nullptr, nullptr, nullptr, nullptr, hid_bf, F);

  // 10) grouped FFN2 (identity rows -> TDM): moe[tok] += gate * (hid @ w2[e] + b2[e])
  k_gemm<3, true><<<dim3(NT, D / 128), 256, 0, stream>>>(
      hid_bf, F, w2_bf, D, F, CAP, nullptr, rtile_e, (long)F * D,
      b2, D, nullptr, slot_sc, slot_tok, moe, nullptr, D);

  // 11) final residual
  k_addres<<<(T * D / 4 + 255) / 256, 256, 0, stream>>>(x2, moe, out, T * D / 4);
}
